// Global_Branch_3951369912399
// MI455X (gfx1250) — matrix operand order repaired and checked by reading, not compiled, not run on hardware
//
#include <hip/hip_runtime.h>
#include <hip/hip_bf16.h>
#include <math.h>

// ---------------------------------------------------------------------------
// CDNA5 (gfx1250) WMMA implementation of the audio/video co-attention model.
// wave32, v_wmma_f32_16x16x32_bf16 matrix cores, f32 accumulate.
// Block tile 64x128, 4 waves, each wave 32x64 (2x4 WMMA tiles).
// ---------------------------------------------------------------------------

typedef __attribute__((ext_vector_type(16))) __bf16 v16bf;
typedef __attribute__((ext_vector_type(8)))  __bf16 v8bf;
typedef __attribute__((ext_vector_type(2)))  __bf16 v2bf;
typedef __attribute__((ext_vector_type(8)))  float  v8f;

#define BM 64
#define BN 128
#define BKT 32
#define LDK 40   // BKT + 8 pad; row pitch 80 bytes keeps 16B alignment per fragment read

// AMODE: 0 = fp32 A, 1 = bf16 A (score matrix), 2 = tanh(concat(A1,A2)) fp32
template<int AMODE, bool RELU>
__global__ __launch_bounds__(128) void gemm_kernel(
    const void* __restrict__ Aptr, const float* __restrict__ A2,
    const float* __restrict__ Bptr, const float* __restrict__ bias,
    float* __restrict__ Cptr,
    int M, int N, int K, int Ksplit,
    long sA, long sB, long sC,
    int inner,
    long bAo, long bAi, long bBo, long bBi, long bCo, long bCi)
{
  __shared__ __align__(16) __bf16 As[BM * LDK];   //  5.0 KiB
  __shared__ __align__(16) __bf16 Bs[BN * LDK];   // 10.0 KiB, transposed: Bs[n][k]

  const int tid  = threadIdx.x;
  const int wave = tid >> 5;
  const int lane = tid & 31;
  const int wm   = (wave >> 1) * 32;
  const int wn   = (wave & 1) * 64;
  const int half = lane >> 4;
  const int l    = lane & 15;

  const int m0 = blockIdx.y * BM;
  const int n0 = blockIdx.x * BN;
  const int z  = blockIdx.z;
  const int zo = z / inner;
  const int zi = z - zo * inner;
  const long offA = (long)zo * bAo + (long)zi * bAi;
  const long offB = (long)zo * bBo + (long)zi * bBi;
  const long offC = (long)zo * bCo + (long)zi * bCi;

  const float*  A1f = (const float*)Aptr;
  const __bf16* A1b = (const __bf16*)Aptr;

  v8f acc[2][4];
  #pragma unroll
  for (int a = 0; a < 2; ++a)
    #pragma unroll
    for (int b = 0; b < 4; ++b)
      #pragma unroll
      for (int j = 0; j < 8; ++j) acc[a][b][j] = 0.0f;

  // A staging coords: 8 consecutive K per thread, 2 row-passes
  const int ar = tid >> 2;          // 0..31
  const int ac = (tid & 3) * 8;     // 0,8,16,24

  for (int k0 = 0; k0 < K; k0 += BKT) {
    // ---- stage A tile (64x32): global_load_b128 x2 -> cvt -> ds_store_b128 --
    #pragma unroll
    for (int pass = 0; pass < 2; ++pass) {
      int r  = ar + pass * 32;
      int gm = m0 + r;
      int gk = k0 + ac;
      v8bf pk;
      if (AMODE == 1) {
        pk = *(const v8bf*)(A1b + offA + (long)gm * sA + gk);
      } else {
        const float* src;
        if (AMODE == 2) {
          // Ksplit is a multiple of BKT, so the whole 8-elem span is one side
          src = (gk < Ksplit) ? (A1f + offA + (long)gm * sA + gk)
                              : (A2  + offA + (long)gm * sA + (gk - Ksplit));
        } else {
          src = A1f + offA + (long)gm * sA + gk;
        }
        float4 f0 = *(const float4*)src;
        float4 f1 = *(const float4*)(src + 4);
        float vv[8] = {f0.x, f0.y, f0.z, f0.w, f1.x, f1.y, f1.z, f1.w};
        #pragma unroll
        for (int j = 0; j < 8; ++j) {
          float x = vv[j];
          if (AMODE == 2) x = tanhf(x);
          pk[j] = (__bf16)x;
        }
      }
      *(v8bf*)&As[r * LDK + ac] = pk;
    }

    // ---- stage B tile (32x128) transposed: 2k x 2n quads per thread --------
    // global_load_b64 x2 -> cvt_pk -> ds_store_b32 x2
    #pragma unroll
    for (int it = 0; it < (BKT / 2) * (BN / 2) / 128; ++it) {   // 8 iters
      int i  = tid + it * 128;
      int n2 = (i & 63) * 2;        // 0..126
      int kk = (i >> 6) * 2;        // 0..30
      const float* src = Bptr + offB + (long)(k0 + kk) * sB + (n0 + n2);
      float2 r0 = *(const float2*)src;          // row kk  : n2, n2+1
      float2 r1 = *(const float2*)(src + sB);   // row kk+1: n2, n2+1
      v2bf p0; p0[0] = (__bf16)r0.x; p0[1] = (__bf16)r1.x;
      v2bf p1; p1[0] = (__bf16)r0.y; p1[1] = (__bf16)r1.y;
      *(v2bf*)&Bs[(n2    ) * LDK + kk] = p0;
      *(v2bf*)&Bs[(n2 + 1) * LDK + kk] = p1;
    }
    __syncthreads();

    // ---- fragments per ISA VGPR layouts ------------------------------------
    // A 16-bit 16x32: lanes 0-15 hold K 0-7 / 16-23; lanes 16-31 hold K 8-15 / 24-31
    // B 16-bit 32x16: lanes 0-15 hold K 0-7 / 16-23 of col N=lane; lanes 16-31 hold K 8-15 / 24-31
    v16bf afrag[2], bfrag[4];
    #pragma unroll
    for (int t = 0; t < 2; ++t) {
      const __bf16* ap = &As[(wm + t * 16 + l) * LDK + half * 8];
      v8bf alo = *(const v8bf*)ap;
      v8bf ahi = *(const v8bf*)(ap + 16);
      #pragma unroll
      for (int j = 0; j < 8; ++j) { afrag[t][j] = alo[j]; afrag[t][8 + j] = ahi[j]; }
    }
    #pragma unroll
    for (int t = 0; t < 4; ++t) {
      const __bf16* bp = &Bs[(wn + t * 16 + l) * LDK + half * 8];
      v8bf blo = *(const v8bf*)bp;
      v8bf bhi = *(const v8bf*)(bp + 16);
      #pragma unroll
      for (int j = 0; j < 8; ++j) { bfrag[t][j] = blo[j]; bfrag[t][8 + j] = bhi[j]; }
    }
    #pragma unroll
    for (int tm = 0; tm < 2; ++tm)
      #pragma unroll
      for (int tn = 0; tn < 4; ++tn)
        acc[tm][tn] = __builtin_amdgcn_wmma_f32_16x16x32_bf16(
            false, afrag[tm], false, bfrag[tn], (short)0, acc[tm][tn], false, false);
    __syncthreads();
  }

  // ---- epilogue: bias (+ReLU), f32 store -----------------------------------
  #pragma unroll
  for (int tm = 0; tm < 2; ++tm) {
    #pragma unroll
    for (int tn = 0; tn < 4; ++tn) {
      int col = n0 + wn + tn * 16 + l;
      float bv = bias ? bias[col] : 0.0f;
      #pragma unroll
      for (int r = 0; r < 8; ++r) {
        int row = m0 + wm + tm * 16 + r + half * 8;
        float v = acc[tm][tn][r] + bv;
        if (RELU) v = fmaxf(v, 0.0f);
        Cptr[offC + (long)row * sC + col] = v;
      }
    }
  }
}

// ---------------------------------------------------------------------------
// sk[b,n,t] = sum_e x[(b*T+t)*D + n*h + e] * w[e]   (h=128, D=512, NH=4, T=256)
// one wave per (b,t,n)
__global__ __launch_bounds__(128) void score_dot_kernel(
    const float* __restrict__ x, const float* __restrict__ w, float* __restrict__ out)
{
  int wave = threadIdx.x >> 5, lane = threadIdx.x & 31;
  int g  = blockIdx.x * 4 + wave;     // (b*T + t)*NH + n
  int n  = g & 3;
  int bt = g >> 2;
  long base = (long)bt * 512 + n * 128;
  float s = 0.0f;
  #pragma unroll
  for (int j = 0; j < 4; ++j) s += x[base + lane + j * 32] * w[lane + j * 32];
  #pragma unroll
  for (int o = 16; o > 0; o >>= 1) s += __shfl_xor(s, o, 32);
  if (lane == 0) {
    int b = bt >> 8;      // / 256
    int t = bt & 255;
    out[(long)(b * 4 + n) * 256 + t] = s;
  }
}

// score[row, k] = softmax_k( tanh(sq[row] + sk[bn, k]) ) -> bf16 (feeds WMMA)
// one wave per (b,n,q) row; Tk = 256
__global__ __launch_bounds__(128) void softmax_kernel(
    const float* __restrict__ sk, const float* __restrict__ sq, __bf16* __restrict__ score)
{
  int wave = threadIdx.x >> 5, lane = threadIdx.x & 31;
  long row = (long)blockIdx.x * 4 + wave;    // (b*NH+n)*Tq + q
  long bn  = row >> 8;
  int  q   = (int)(row & 255);
  float sqv = sq[bn * 256 + q];
  float v[8];
  float m = -1e30f;
  #pragma unroll
  for (int j = 0; j < 8; ++j) {
    int k = lane + j * 32;
    float x = tanhf(sqv + sk[bn * 256 + k]);
    v[j] = x; m = fmaxf(m, x);
  }
  #pragma unroll
  for (int o = 16; o > 0; o >>= 1) m = fmaxf(m, __shfl_xor(m, o, 32));
  float s = 0.0f;
  #pragma unroll
  for (int j = 0; j < 8; ++j) { v[j] = __expf(v[j] - m); s += v[j]; }
  #pragma unroll
  for (int o = 16; o > 0; o >>= 1) s += __shfl_xor(s, o, 32);
  float inv = 1.0f / s;
  #pragma unroll
  for (int j = 0; j < 8; ++j) {
    int k = lane + j * 32;
    score[row * 256 + k] = (__bf16)(v[j] * inv);
  }
}

// LayerNorm over D=512, optional fused residual add; one wave per row
template<bool ADD>
__global__ __launch_bounds__(128) void ln_kernel(
    const float* __restrict__ x, const float* __restrict__ res,
    const float* __restrict__ g, const float* __restrict__ bta,
    float* __restrict__ out, float eps)
{
  int wave = threadIdx.x >> 5, lane = threadIdx.x & 31;
  long row  = (long)blockIdx.x * 4 + wave;
  long base = row * 512;
  float v[16]; float s = 0.0f;
  #pragma unroll
  for (int j = 0; j < 16; ++j) {
    int i = lane + j * 32;
    float t = x[base + i];
    if (ADD) t += res[base + i];
    v[j] = t; s += t;
  }
  #pragma unroll
  for (int o = 16; o > 0; o >>= 1) s += __shfl_xor(s, o, 32);
  float mean = s * (1.0f / 512.0f);
  float vs = 0.0f;
  #pragma unroll
  for (int j = 0; j < 16; ++j) { float c = v[j] - mean; v[j] = c; vs += c * c; }
  #pragma unroll
  for (int o = 16; o > 0; o >>= 1) vs += __shfl_xor(vs, o, 32);
  float rstd = rsqrtf(vs * (1.0f / 512.0f) + eps);
  #pragma unroll
  for (int j = 0; j < 16; ++j) {
    int i = lane + j * 32;
    out[base + i] = v[j] * rstd * g[i] + bta[i];
  }
}

__global__ void avg_kernel(const float4* __restrict__ a, const float4* __restrict__ b,
                           float4* __restrict__ o, long n4)
{
  long i = (long)blockIdx.x * blockDim.x + threadIdx.x;
  if (i < n4) {
    float4 x = a[i], y = b[i];
    float4 r;
    r.x = 0.5f * (x.x + y.x); r.y = 0.5f * (x.y + y.y);
    r.z = 0.5f * (x.z + y.z); r.w = 0.5f * (x.w + y.w);
    o[i] = r;
  }
}

// ---------------------------------------------------------------------------
// Host orchestration
// ---------------------------------------------------------------------------
namespace {

struct Att { const float *bk,*bp,*bq,*w,*wk,*wp,*wq; };
struct Ffn { const float *b,*b1,*b2,*bd,*g,*w1,*w2,*wd; };

constexpr long Bsz = 128, T = 256, D = 512, NHc = 4, Hh = 128, Hid = 1024;
constexpr long Mrows = Bsz * T;                 // 32768
constexpr long MD = Mrows * D;                  // 16,777,216 floats (64 MiB)

static void launch_gemm(hipStream_t st, int amode, bool relu,
    const void* A, const float* A2, const float* W, const float* bias, float* C,
    int M, int N, int K, int Ksplit, long sA, long sB, long sC,
    int inner, long bAo, long bAi, long bBo, long bBi, long bCo, long bCi, int batch)
{
  dim3 grid(N / BN, M / BM, batch), blk(128);
  if (amode == 0) {
    if (relu) gemm_kernel<0,true ><<<grid, blk, 0, st>>>(A, A2, W, bias, C, M, N, K, Ksplit, sA, sB, sC, inner, bAo, bAi, bBo, bBi, bCo, bCi);
    else      gemm_kernel<0,false><<<grid, blk, 0, st>>>(A, A2, W, bias, C, M, N, K, Ksplit, sA, sB, sC, inner, bAo, bAi, bBo, bBi, bCo, bCi);
  } else if (amode == 1) {
    gemm_kernel<1,false><<<grid, blk, 0, st>>>(A, A2, W, bias, C, M, N, K, Ksplit, sA, sB, sC, inner, bAo, bAi, bBo, bBi, bCo, bCi);
  } else {
    if (relu) gemm_kernel<2,true ><<<grid, blk, 0, st>>>(A, A2, W, bias, C, M, N, K, Ksplit, sA, sB, sC, inner, bAo, bAi, bBo, bBi, bCo, bCi);
    else      gemm_kernel<2,false><<<grid, blk, 0, st>>>(A, A2, W, bias, C, M, N, K, Ksplit, sA, sB, sC, inner, bAo, bAi, bBo, bBi, bCo, bCi);
  }
}

} // namespace

extern "C" void kernel_launch(void* const* d_in, const int* in_sizes, int n_in,
                              void* d_out, int out_size, void* d_ws, size_t ws_size,
                              hipStream_t stream)
{
  (void)in_sizes; (void)n_in; (void)out_size; (void)ws_size;
  const float* audio = (const float*)d_in[0];
  const float* video = (const float*)d_in[1];

  // params flattened in JAX pytree (sorted-key) order
  int i = 2;
  auto getAtt = [&](Att& a) {
    a.bk = (const float*)d_in[i++]; a.bp = (const float*)d_in[i++];
    a.bq = (const float*)d_in[i++]; a.w  = (const float*)d_in[i++];
    a.wk = (const float*)d_in[i++]; a.wp = (const float*)d_in[i++];
    a.wq = (const float*)d_in[i++];
  };
  auto getFfn = [&](Ffn& f) {
    f.b  = (const float*)d_in[i++]; f.b1 = (const float*)d_in[i++];
    f.b2 = (const float*)d_in[i++]; f.bd = (const float*)d_in[i++];
    f.g  = (const float*)d_in[i++]; f.w1 = (const float*)d_in[i++];
    f.w2 = (const float*)d_in[i++]; f.wd = (const float*)d_in[i++];
  };
  Att att_a2v;   getAtt(att_a2v);
  Att att_audio; getAtt(att_audio);
  Att att_v2a;   getAtt(att_v2a);
  Att att_video; getAtt(att_video);
  Ffn ffn_audio; getFfn(ffn_audio);
  Ffn ffn_video; getFfn(ffn_video);
  Att sa_a_att;  getAtt(sa_a_att);
  const float* saA_b = (const float*)d_in[i++];
  const float* saA_g = (const float*)d_in[i++];
  Att sa_v_att;  getAtt(sa_v_att);
  const float* saV_b = (const float*)d_in[i++];
  const float* saV_g = (const float*)d_in[i++];

  // workspace slots (64 MiB each)
  float* ws = (float*)d_ws;
  float* Aslot = ws + 0 * MD;
  float* Vslot = ws + 1 * MD;
  float* ETA   = ws + 2 * MD;
  float* ETV   = ws + 3 * MD;
  float* KX    = ws + 4 * MD;
  float* QX    = ws + 5 * MD;
  __bf16* SCORE = (__bf16*)(ws + 6 * MD);       // 33.55M bf16 = one slot
  float* O1 = ws + 7 * MD;
  float* O2 = ws + 8 * MD;
  float* O4 = ws + 9 * MD;
  float* XD = O4;                               // reused by FFN
  float* H1 = KX;                               // spans slots 4+5 ([M,1024])
  float* X2 = (float*)SCORE;                    // score slot free during FFN
  float* SK = ws + 10 * MD;
  float* SQ = SK + (Bsz * NHc * T);             // 131072 floats each

  const int LN_GRID = (int)(Mrows / 4);         // wave per row, 4 waves/block
  const int SD_GRID = (int)(Bsz * T * NHc / 4);
  const int SM_GRID = (int)(Bsz * NHc * T / 4);

  auto run_attn = [&](const float* Kin, const float* Qin, const Att& p, float* out) {
    // kx = K @ wk + bk ; qx = Q @ wq + bq
    launch_gemm(stream, 0, false, Kin, nullptr, p.wk, p.bk, KX,
                (int)Mrows, (int)D, (int)D, 0, D, D, D, 1, 0,0,0,0,0,0, 1);
    launch_gemm(stream, 0, false, Qin, nullptr, p.wq, p.bq, QX,
                (int)Mrows, (int)D, (int)D, 0, D, D, D, 1, 0,0,0,0,0,0, 1);
    score_dot_kernel<<<SD_GRID, 128, 0, stream>>>(KX, p.w,       SK);
    score_dot_kernel<<<SD_GRID, 128, 0, stream>>>(QX, p.w + Hh,  SQ);
    softmax_kernel<<<SM_GRID, 128, 0, stream>>>(SK, SQ, SCORE);
    // out[b,q,n,e] = sum_k score[b,n,q,k] * kx[b,k,n,e]  (batched over b*NH, into QX)
    launch_gemm(stream, 1, false, SCORE, nullptr, KX, nullptr, QX,
                (int)T, (int)Hh, (int)T, 0, /*sA*/T, /*sB*/D, /*sC*/D,
                /*inner*/(int)NHc,
                /*bAo*/NHc * T * T, /*bAi*/T * T,
                /*bBo*/T * D,       /*bBi*/Hh,
                /*bCo*/T * D,       /*bCi*/Hh,
                (int)(Bsz * NHc));
    // proj: out = attn_out @ wp + bp
    launch_gemm(stream, 0, false, QX, nullptr, p.wp, p.bp, out,
                (int)Mrows, (int)D, (int)D, 0, D, D, D, 1, 0,0,0,0,0,0, 1);
  };

  auto run_ffn = [&](const float* it, const float* etOld, const Ffn& p, float* etOut) {
    // xd = relu(tanh(concat(it, etOld)) @ wd + bd)
    launch_gemm(stream, 2, true, it, etOld, p.wd, p.bd, XD,
                (int)Mrows, (int)D, (int)(2 * D), (int)D, D, D, D, 1, 0,0,0,0,0,0, 1);
    ln_kernel<false><<<LN_GRID, 128, 0, stream>>>(XD, nullptr, p.g, p.b, XD, 1e-6f);
    // h1 = relu(x @ w1 + b1)
    launch_gemm(stream, 0, true, XD, nullptr, p.w1, p.b1, H1,
                (int)Mrows, (int)Hid, (int)D, 0, D, Hid, Hid, 1, 0,0,0,0,0,0, 1);
    // x2 = h1 @ w2 + b2
    launch_gemm(stream, 0, false, H1, nullptr, p.w2, p.b2, X2,
                (int)Mrows, (int)D, (int)Hid, 0, Hid, D, D, 1, 0,0,0,0,0,0, 1);
    ln_kernel<true><<<LN_GRID, 128, 0, stream>>>(X2, XD, p.g, p.b, etOut, 1e-6f);
  };

  // ---- self attention -----------------------------------------------------
  run_attn(audio, audio, sa_a_att, O1);
  ln_kernel<true><<<LN_GRID, 128, 0, stream>>>(O1, audio, saA_g, saA_b, Aslot, 1e-5f);
  run_attn(video, video, sa_v_att, O1);
  ln_kernel<true><<<LN_GRID, 128, 0, stream>>>(O1, video, saV_g, saV_b, Vslot, 1e-5f);

  const float* etA = Aslot;
  const float* etV = Vslot;

  // ---- hops ---------------------------------------------------------------
  for (int hop = 0; hop < 2; ++hop) {
    run_attn(Aslot, etA, att_audio, O1);
    run_attn(Aslot, etV, att_v2a,   O2);
    avg_kernel<<<(int)(MD / 4 / 256), 256, 0, stream>>>(
        (const float4*)O1, (const float4*)O2, (float4*)O1, MD / 4);   // it_a
    run_attn(Vslot, etV, att_video, O2);
    run_attn(Vslot, etA, att_a2v,   O4);
    avg_kernel<<<(int)(MD / 4 / 256), 256, 0, stream>>>(
        (const float4*)O2, (const float4*)O4, (float4*)O2, MD / 4);   // it_v
    run_ffn(O1, etA, ffn_audio, ETA);
    run_ffn(O2, etV, ffn_video, ETV);
    etA = ETA; etV = ETV;
  }

  hipMemcpyAsync(d_out, ETA, MD * sizeof(float), hipMemcpyDeviceToDevice, stream);
  hipMemcpyAsync((float*)d_out + MD, ETV, MD * sizeof(float), hipMemcpyDeviceToDevice, stream);
}
